// SpikingAttention_83485574300331
// MI455X (gfx1250) — matrix operand order repaired and checked by reading, not compiled, not run on hardware
//
#include <hip/hip_runtime.h>
#include <hip/hip_bf16.h>

#define BATCH   16
#define SEQ     2048
#define CH      768
#define KSTEP   32
#define LDS_STR (CH + 8)          // +8 bf16 pad -> conflict-free ds_load_b128
#define EPS_BN  1e-5f

typedef __bf16 bf16x16 __attribute__((ext_vector_type(16)));
typedef float  f32x8   __attribute__((ext_vector_type(8)));

struct Bits256 { uint4 lo, hi; };   // 32 bytes == bf16x16

__device__ __forceinline__ unsigned short f2bf(float f) {
    union { float f; unsigned u; } v; v.f = f;
    unsigned r = v.u + 0x7FFFu + ((v.u >> 16) & 1u);   // round-to-nearest-even
    return (unsigned short)(r >> 16);
}

__device__ __forceinline__ bf16x16 make_frag(uint4 lo, uint4 hi) {
    Bits256 t; t.lo = lo; t.hi = hi;
    return __builtin_bit_cast(bf16x16, t);
}

// ---------------- k0: weights fp32 -> bf16, zero the score-column accumulator
__global__ void __launch_bounds__(256)
k0_prep(const float* __restrict__ Wq, const float* __restrict__ Wk,
        const float* __restrict__ Wv,
        unsigned short* __restrict__ Wqb, unsigned short* __restrict__ Wkb,
        unsigned short* __restrict__ Wvb, float* __restrict__ w_accum) {
    int idx = blockIdx.x * 256 + threadIdx.x;
    if (idx < CH * CH) {
        Wqb[idx] = f2bf(Wq[idx]);
        Wkb[idx] = f2bf(Wk[idx]);
        Wvb[idx] = f2bf(Wv[idx]);
    }
    if (idx < BATCH * SEQ) w_accum[idx] = 0.0f;
}

// ---------------- k1: fused Q/K/V projection via v_wmma_f32_16x16x32_bf16
// One workgroup (8 waves) owns a 16-row block of x staged once into LDS as bf16;
// the 8 waves split the 3*48 = 144 output column tiles (18 tiles each).
__global__ void __launch_bounds__(256)
k1_qkv(const float* __restrict__ x,
       const unsigned short* __restrict__ Wqb, const float* __restrict__ bq,
       const unsigned short* __restrict__ Wkb, const float* __restrict__ bk,
       const unsigned short* __restrict__ Wvb, const float* __restrict__ bv,
       unsigned short* __restrict__ Qb, unsigned short* __restrict__ Kb,
       float* __restrict__ Vf)
{
    __shared__ unsigned short xs[16 * LDS_STR];
    const int tid  = threadIdx.x;
    const int row0 = blockIdx.x * 16;

    // stage 16x768 fp32 -> bf16 into LDS (A-matrix source, row major, padded)
    for (int m = 0; m < 16; ++m)
        for (int c = tid; c < CH; c += 256)
            xs[m * LDS_STR + c] = f2bf(x[(size_t)(row0 + m) * CH + c]);
    __syncthreads();

    const int lane = tid & 31;
    const int wave = tid >> 5;
    const int m = lane & 15;       // A row / B-D column within tile
    const int h = lane >> 4;       // lane-half

    for (int t = 0; t < 18; ++t) {
        const int tile = wave * 18 + t;
        const int mat  = tile / 48;        // 0=Q 1=K 2=V
        const int col0 = (tile % 48) * 16;
        const unsigned short* W  = (mat == 0) ? Wqb : (mat == 1) ? Wkb : Wvb;
        const float*          bs = (mat == 0) ? bq  : (mat == 1) ? bk  : bv;

        f32x8 acc = {};
        for (int kd = 0; kd < CH; kd += KSTEP) {
            // A 16x32: lane holds K = kd+16h..+15, contiguous (from LDS)
            const unsigned short* ap = &xs[m * LDS_STR + kd + 16 * h];
            bf16x16 a = make_frag(*(const uint4*)ap, *(const uint4*)(ap + 8));
            // B 32x16: column n=m holds K = kd+16h..+15 == W row (col0+m), contiguous
            const unsigned short* bp = W + (size_t)(col0 + m) * CH + kd + 16 * h;
            bf16x16 bf = make_frag(*(const uint4*)bp, *(const uint4*)(bp + 8));
            acc = __builtin_amdgcn_wmma_f32_16x16x32_bf16(
                      false, a, false, bf, (short)0, acc, false, false);
        }
        const float bb   = bs[col0 + m];
        const int rbase  = row0 + 8 * h;    // D: VGPR r -> row r + 8h
        if (mat == 2) {
            #pragma unroll
            for (int r = 0; r < 8; ++r)
                Vf[(size_t)(rbase + r) * CH + col0 + m] = acc[r] + bb;
        } else {
            unsigned short* O = (mat == 0) ? Qb : Kb;
            #pragma unroll
            for (int r = 0; r < 8; ++r)
                O[(size_t)(rbase + r) * CH + col0 + m] = f2bf(acc[r] + bb);
        }
    }
}

// ---------------- k2: w[b,k] = sum_q sigmoid(scale * Q[b,q].K[b,k])
// Wave computes a 16q x 64k strip (4 WMMA accumulators, A reused 4x per K-step),
// applies sigmoid, reduces columns over the 16 queries, atomically accumulates.
__global__ void __launch_bounds__(256)
k2_scores(const unsigned short* __restrict__ Qb,
          const unsigned short* __restrict__ Kb,
          float* __restrict__ w_accum)
{
    const int tid  = threadIdx.x;
    const int lane = tid & 31;
    const int wave = tid >> 5;
    const int m = lane & 15, h = lane >> 4;
    const int b  = blockIdx.z;
    const int q0 = blockIdx.x * 16;
    const int k0 = blockIdx.y * 512 + wave * 64;

    const unsigned short* qp = Qb + (size_t)(b * SEQ + q0 + m) * CH;
    const unsigned short* kp = Kb + (size_t)(b * SEQ + k0 + m) * CH;

    f32x8 a0 = {}, a1 = {}, a2 = {}, a3 = {};
    for (int kd = 0; kd < CH; kd += KSTEP) {
        const unsigned short* ap = qp + kd + 16 * h;
        bf16x16 a = make_frag(*(const uint4*)ap, *(const uint4*)(ap + 8));
        const unsigned short* bp = kp + kd + 16 * h;
        bf16x16 b0 = make_frag(*(const uint4*)(bp),           *(const uint4*)(bp + 8));
        bf16x16 b1 = make_frag(*(const uint4*)(bp + 16 * CH), *(const uint4*)(bp + 16 * CH + 8));
        bf16x16 b2 = make_frag(*(const uint4*)(bp + 32 * CH), *(const uint4*)(bp + 32 * CH + 8));
        bf16x16 b3 = make_frag(*(const uint4*)(bp + 48 * CH), *(const uint4*)(bp + 48 * CH + 8));
        a0 = __builtin_amdgcn_wmma_f32_16x16x32_bf16(false, a, false, b0, (short)0, a0, false, false);
        a1 = __builtin_amdgcn_wmma_f32_16x16x32_bf16(false, a, false, b1, (short)0, a1, false, false);
        a2 = __builtin_amdgcn_wmma_f32_16x16x32_bf16(false, a, false, b2, (short)0, a2, false, false);
        a3 = __builtin_amdgcn_wmma_f32_16x16x32_bf16(false, a, false, b3, (short)0, a3, false, false);
    }

    const float scl = 0.03608439182435161f;   // 768^-0.5
    float* wdst = w_accum + (size_t)b * SEQ + k0;

#define COLSUM(ACC, J)                                                     \
    {                                                                      \
        float s = 0.0f;                                                    \
        _Pragma("unroll")                                                  \
        for (int r = 0; r < 8; ++r)                                        \
            s += 1.0f / (1.0f + __expf(-scl * ACC[r]));                    \
        s += __shfl_xor(s, 16, 32);   /* join rows 0-7 with rows 8-15 */   \
        if (lane < 16) atomicAdd(&wdst[(J) * 16 + lane], s);               \
    }
    COLSUM(a0, 0) COLSUM(a1, 1) COLSUM(a2, 2) COLSUM(a3, 3)
#undef COLSUM
}

// ---------------- k3: out_pre[b,d] = (1/T) * sum_k w[b,k] * V[b,k,d]
__global__ void __launch_bounds__(256)
k3_attend(const float* __restrict__ Vf, const float* __restrict__ w_accum,
          float* __restrict__ out_pre)
{
    const int d = blockIdx.x * 256 + threadIdx.x;   // 0..767
    const int b = blockIdx.y;
    const float* wp = w_accum + (size_t)b * SEQ;
    const float* vp = Vf + (size_t)b * SEQ * CH + d;
    float s0 = 0.f, s1 = 0.f, s2 = 0.f, s3 = 0.f;
    for (int k = 0; k < SEQ; k += 4) {
        s0 = fmaf(wp[k + 0], vp[(size_t)(k + 0) * CH], s0);
        s1 = fmaf(wp[k + 1], vp[(size_t)(k + 1) * CH], s1);
        s2 = fmaf(wp[k + 2], vp[(size_t)(k + 2) * CH], s2);
        s3 = fmaf(wp[k + 3], vp[(size_t)(k + 3) * CH], s3);
    }
    out_pre[b * CH + d] = ((s0 + s1) + (s2 + s3)) * (1.0f / SEQ);
}

// ---------------- k4: BatchNorm1d over the batch dimension (training stats)
__global__ void __launch_bounds__(256)
k4_bn(const float* __restrict__ out_pre, const float* __restrict__ gamma,
      const float* __restrict__ beta, float* __restrict__ out)
{
    const int d = blockIdx.x * 256 + threadIdx.x;
    if (d >= CH) return;
    float v[BATCH], mu = 0.f, var = 0.f;
    #pragma unroll
    for (int b = 0; b < BATCH; ++b) { v[b] = out_pre[b * CH + d]; mu += v[b]; }
    mu *= (1.0f / BATCH);
    #pragma unroll
    for (int b = 0; b < BATCH; ++b) { float t = v[b] - mu; var += t * t; }
    var *= (1.0f / BATCH);                       // biased variance
    const float g = rsqrtf(var + EPS_BN) * gamma[d];
    const float bt = beta[d];
    #pragma unroll
    for (int b = 0; b < BATCH; ++b) out[b * CH + d] = (v[b] - mu) * g + bt;
}

extern "C" void kernel_launch(void* const* d_in, const int* in_sizes, int n_in,
                              void* d_out, int out_size, void* d_ws, size_t ws_size,
                              hipStream_t stream) {
    const float* x  = (const float*)d_in[0];
    const float* Wq = (const float*)d_in[1];
    const float* bq = (const float*)d_in[2];
    const float* Wk = (const float*)d_in[3];
    const float* bk = (const float*)d_in[4];
    const float* Wv = (const float*)d_in[5];
    const float* bv = (const float*)d_in[6];
    const float* gamma = (const float*)d_in[7];
    const float* beta  = (const float*)d_in[8];
    float* out = (float*)d_out;

    char* ws = (char*)d_ws;
    size_t off = 0;
    auto carve = [&](size_t bytes) -> void* {
        void* p = ws + off;
        off += (bytes + 255) & ~(size_t)255;
        return p;
    };
    unsigned short* Wqb = (unsigned short*)carve((size_t)CH * CH * 2);
    unsigned short* Wkb = (unsigned short*)carve((size_t)CH * CH * 2);
    unsigned short* Wvb = (unsigned short*)carve((size_t)CH * CH * 2);
    unsigned short* Qb  = (unsigned short*)carve((size_t)BATCH * SEQ * CH * 2);
    unsigned short* Kb  = (unsigned short*)carve((size_t)BATCH * SEQ * CH * 2);
    float*          Vf  = (float*)carve((size_t)BATCH * SEQ * CH * 4);
    float*          wac = (float*)carve((size_t)BATCH * SEQ * 4);
    float*          opre= (float*)carve((size_t)BATCH * CH * 4);

    k0_prep<<<(CH * CH + 255) / 256, 256, 0, stream>>>(Wq, Wk, Wv, Wqb, Wkb, Wvb, wac);

    k1_qkv<<<(BATCH * SEQ) / 16, 256, 0, stream>>>(x, Wqb, bq, Wkb, bk, Wvb, bv,
                                                   Qb, Kb, Vf);

    dim3 g2(SEQ / 16, SEQ / 512, BATCH);          // 128 x 4 x 16
    k2_scores<<<g2, 256, 0, stream>>>(Qb, Kb, wac);

    dim3 g3(CH / 256, BATCH);                     // 3 x 16
    k3_attend<<<g3, 256, 0, stream>>>(Vf, wac, opre);

    k4_bn<<<(CH + 255) / 256, 256, 0, stream>>>(opre, gamma, beta, out);
}
